// MultiHeadAttention_5738076308266
// MI455X (gfx1250) — matrix operand order repaired and checked by reading, not compiled, not run on hardware
//
#include <hip/hip_runtime.h>

// Multi-head attention forward, MI455X (gfx1250, wave32, WMMA, TDM, ASYNCcnt).
// B=2, S=2048, D=1024, H=16, DK=64.
// GEMMs: v_wmma_f32_16x16x32_bf16 (fp32 accumulate). Softmax fp32 (v_exp_f32).
// proj_kernel stages W tiles via TENSOR_LOAD_TO_LDS (TDM, hardware LDS padding);
// outproj_kernel stages W tiles via GLOBAL_LOAD_ASYNC_TO_LDS_B128 (ASYNCcnt).
// d_out = [out (B,S,D) fp32 | attn_mean (B,S,S) fp32], ws = Q|K|Vt|ctx bf16 (32MB).

constexpr int kBn  = 2;
constexpr int kSn  = 2048;
constexpr int kDn  = 1024;
constexpr int kHn  = 16;
constexpr int kDK  = 64;
constexpr int kTok = kBn * kSn;   // 4096 tokens

constexpr int kWP  = 36;          // padded LDS row stride (floats): 16 lanes -> 16 banks

typedef __attribute__((ext_vector_type(16))) __bf16 v16bf;
typedef __attribute__((ext_vector_type(8)))  __bf16 v8bf;
typedef __attribute__((ext_vector_type(8)))  float  v8f;

typedef int v4i __attribute__((vector_size(16)));
typedef __attribute__((address_space(1))) v4i as1_v4i;   // global 16B vector
typedef __attribute__((address_space(3))) v4i as3_v4i;   // LDS 16B vector
typedef __attribute__((address_space(3))) const void as3c_void;

typedef unsigned int u32x4 __attribute__((vector_size(16)));
typedef int          i32x4 __attribute__((vector_size(16)));
typedef int          i32x8 __attribute__((vector_size(32)));

union V16 { v16bf v; v8bf h2[2]; };

static __device__ __forceinline__ void cvt8(V16& d, int base, float4 a, float4 b) {
  d.v[base + 0] = (__bf16)a.x; d.v[base + 1] = (__bf16)a.y;
  d.v[base + 2] = (__bf16)a.z; d.v[base + 3] = (__bf16)a.w;
  d.v[base + 4] = (__bf16)b.x; d.v[base + 5] = (__bf16)b.y;
  d.v[base + 6] = (__bf16)b.z; d.v[base + 7] = (__bf16)b.w;
}

// one 16-byte async global->LDS copy (tracked by ASYNCcnt)
static __device__ __forceinline__ void async_cp16(const float* g, float* l) {
  __builtin_amdgcn_global_load_async_to_lds_b128((as1_v4i*)g, (as3_v4i*)l, 0, 0);
}

// workgroup-relative LDS byte address of a __shared__ object
static __device__ __forceinline__ unsigned lds_addr_of(const void* p) {
  return (unsigned)(size_t)(as3c_void*)p;
}

// TDM: DMA one 64x32 fp32 tile (row stride kDn) global->LDS, with hardware
// padding of 4 DWORDs after every 32 DWORDs -> LDS row stride kWP=36 floats.
// Tracked on TENSORcnt. Issue from a single wave.
static __device__ __forceinline__ void tdm_load_tile(const float* gsrc, float* ldst) {
  const size_t ga = (size_t)gsrc;
  const u32x4 g0 = {
    1u,                                               // count=1, user descriptor
    lds_addr_of(ldst),                                // lds_addr (bytes)
    (unsigned)(ga & 0xFFFFFFFFu),                     // global_addr[31:0]
    (unsigned)((ga >> 32) & 0x01FFFFFFu) | (2u << 30) // addr[56:32] | type=2
  };
  const i32x8 g1 = {
    (2 << 16) | (1 << 20) | (4 << 22) | (3 << 25),    // data_size=4B, pad_en,
                                                      // interval=32DW, amount=4DW
    (int)(1024u << 16),                               // tensor_dim0=1024 (lo16)
    (int)(64u  << 16),                                // dim0 hi | tensor_dim1=64
    (int)(32u  << 16),                                // dim1 hi | tile_dim0=32
    64,                                               // tile_dim1=64 | tile_dim2=0
    1024,                                             // tensor_dim0_stride lo32
    0, 0                                              // stride0 hi | stride1
  };
  const i32x4 z4 = {0, 0, 0, 0};
  const i32x8 z8 = {0, 0, 0, 0, 0, 0, 0, 0};
  __builtin_amdgcn_tensor_load_to_lds(g0, g1, z4, z4, z8, 0);
}

// ---------------------------------------------------------------------------
// Stage 1: Y = X @ W^T + bias  (X fp32 NxD, W fp32 DxD torch-Linear layout).
// Block: 128 rows x 64 features, 8 waves; wave w owns rows [16w,16w+16) and
// all 4 feature n-tiles (A-fragment reuse x4). The 64x32 fp32 W chunk is
// TDM-staged into double-buffered padded LDS and shared by all 8 waves.
// transposeOut==0: bf16 out as (B,H,S,DK);  ==1: (B,H,DK,S) (pre-transposed V)
// ---------------------------------------------------------------------------
__global__ __launch_bounds__(256)
void proj_kernel(const float* __restrict__ X, const float* __restrict__ W,
                 const float* __restrict__ bias, __bf16* __restrict__ Y,
                 int transposeOut)
{
  __shared__ float wl[2][64 * kWP];              // 2 x 9 KB, TDM-padded layout

  const int tid  = threadIdx.x;
  const int lane = tid & 31;
  const int wv   = tid >> 5;
  const int l16  = lane & 15;
  const int hi   = lane >> 4;
  const int rb   = (blockIdx.x >> 4) * 128;      // 32 row blocks
  const int eb   = (blockIdx.x & 15) * 64;       // 16 feature blocks

  const float* xrow  = X + (size_t)(rb + wv * 16 + l16) * kDn;  // A row m=l16
  const float* wbase = W + (size_t)eb * kDn;

  v8f acc[4] = {{}, {}, {}, {}};
  if (wv == 0) tdm_load_tile(wbase, wl[0]);      // prime buffer 0
  for (int c = 0; c < 32; ++c) {
    const int kk = c * 32;
    if (wv == 0) {
      if (c + 1 < 32) {
        tdm_load_tile(wbase + kk + 32, wl[(c + 1) & 1]);
        __builtin_amdgcn_s_wait_tensorcnt(1);    // tile c landed (in-order)
      } else {
        __builtin_amdgcn_s_wait_tensorcnt(0);
      }
    }
    __syncthreads();                             // publish LDS tile to all waves

    // A fragment (global, fp32 -> bf16): contiguous K run 16h..16h+15
    V16 a;
    {
      const float4* p0 = reinterpret_cast<const float4*>(xrow + kk + 16 * hi);
      const float4* p1 = reinterpret_cast<const float4*>(xrow + kk + 16 * hi + 8);
      cvt8(a, 0, p0[0], p0[1]);
      cvt8(a, 8, p1[0], p1[1]);
    }
    const float* wbuf = wl[c & 1];
#pragma unroll
    for (int nt = 0; nt < 4; ++nt) {
      // B fragment from LDS: W row n = nt*16+l16, contiguous K run 16h..16h+15
      const float4* q0 =
          reinterpret_cast<const float4*>(wbuf + (nt * 16 + l16) * kWP + 16 * hi);
      V16 bf;
      cvt8(bf, 0, q0[0], q0[1]);
      cvt8(bf, 8, q0[2], q0[3]);
      acc[nt] = __builtin_amdgcn_wmma_f32_16x16x32_bf16(false, a.v, false, bf.v,
                                                        (short)0, acc[nt],
                                                        false, false);
    }
    __syncthreads();                             // done reading before rewrite
  }

#pragma unroll
  for (int nt = 0; nt < 4; ++nt) {
    const int   e  = eb + nt * 16 + l16;
    const float bn = bias[e];
    const int   hh = e >> 6;
    const int   dk = e & 63;
#pragma unroll
    for (int r = 0; r < 8; ++r) {
      const int t  = rb + wv * 16 + r + 8 * hi;   // token
      const int bb = t >> 11;
      const int s  = t & (kSn - 1);
      size_t addr;
      if (!transposeOut)
        addr = (((size_t)(bb * kHn + hh) * kSn) + s) * kDK + dk;   // (B,H,S,DK)
      else
        addr = (((size_t)(bb * kHn + hh) * kDK) + dk) * kSn + s;   // (B,H,DK,S)
      Y[addr] = (__bf16)(acc[nt][r] + bn);
    }
  }
}

// ---------------------------------------------------------------------------
// Stage 2: attention. One block per (b, 16-row query tile), 8 waves.
// Loops over heads; full 16x2048 fp32 score block lives in 128KB LDS.
// Waves 0-3 do P.V, waves 4-7 stream the head-mean of P to attn_weights
// (owner-thread RMW; h==0 overwrites -> replay-safe, no atomics).
// ---------------------------------------------------------------------------
__global__ __launch_bounds__(256)
void attn_kernel(const __bf16* __restrict__ Q, const __bf16* __restrict__ K,
                 const __bf16* __restrict__ Vt, __bf16* __restrict__ ctx,
                 float* __restrict__ attnOut)
{
  __shared__ float sc[16][kSn];                  // 128 KB

  const int lane  = threadIdx.x & 31;
  const int wv    = threadIdx.x >> 5;
  const int l16   = lane & 15;
  const int hi    = lane >> 4;
  const int b     = blockIdx.x >> 7;             // /128
  const int qt    = blockIdx.x & 127;
  const int qbase = qt << 4;

  for (int h = 0; h < kHn; ++h) {
    const size_t bh = (size_t)(b * kHn + h);

    // ---- phase 1: scores = Q.K^T / 8; wave w owns key cols [256w, 256w+256)
    {
      const __bf16* qrow = Q + (bh * kSn + qbase + l16) * kDK;
      V16 aq[2];
#pragma unroll
      for (int c = 0; c < 2; ++c) {
        aq[c].h2[0] = *reinterpret_cast<const v8bf*>(qrow + c * 32 + 16 * hi);
        aq[c].h2[1] = *reinterpret_cast<const v8bf*>(qrow + c * 32 + 16 * hi + 8);
      }
      const int colbase0 = wv * 256;
      for (int ct = 0; ct < 16; ++ct) {
        const int ncb = colbase0 + ct * 16;
        const __bf16* krow = K + (bh * kSn + ncb + l16) * kDK;
        v8f acc = {};
#pragma unroll
        for (int c = 0; c < 2; ++c) {
          V16 bk;
          bk.h2[0] = *reinterpret_cast<const v8bf*>(krow + c * 32 + 16 * hi);
          bk.h2[1] = *reinterpret_cast<const v8bf*>(krow + c * 32 + 16 * hi + 8);
          acc = __builtin_amdgcn_wmma_f32_16x16x32_bf16(false, aq[c].v, false,
                                                        bk.v, (short)0, acc,
                                                        false, false);
        }
#pragma unroll
        for (int r = 0; r < 8; ++r)
          sc[r + 8 * hi][ncb + l16] = acc[r] * 0.125f;   // 1/sqrt(64)
      }
    }
    __syncthreads();

    // ---- phase 2: softmax, one row per half-wave (16 lanes x 128 cols each)
    {
      const int row = (threadIdx.x >> 4) & 15;
      const int j   = threadIdx.x & 15;
      float mx = -3.0e38f;
      for (int i = 0; i < 128; ++i) mx = fmaxf(mx, sc[row][j + 16 * i]);
#pragma unroll
      for (int o = 8; o >= 1; o >>= 1) mx = fmaxf(mx, __shfl_xor(mx, o, 16));
      float sum = 0.f;
      for (int i = 0; i < 128; ++i) {
        const int cidx = j + 16 * i;
        const float e = __expf(sc[row][cidx] - mx);
        sc[row][cidx] = e;
        sum += e;
      }
#pragma unroll
      for (int o = 8; o >= 1; o >>= 1) sum += __shfl_xor(sum, o, 16);
      const float inv = 1.0f / sum;
      for (int i = 0; i < 128; ++i) sc[row][j + 16 * i] *= inv;
    }
    __syncthreads();

    // ---- phase 3: waves 0-3: ctx = P.V (16x64); waves 4-7: mean-P output
    if (wv < 4) {
      v8f acc = {};
      const __bf16* vrow = Vt + (bh * kDK + wv * 16 + l16) * kSn;
      for (int kc = 0; kc < kSn; kc += 32) {
        V16 a;
#pragma unroll
        for (int i = 0; i < 8; ++i) {
          a.v[i]     = (__bf16)sc[l16][kc + 16 * hi + i];
          a.v[8 + i] = (__bf16)sc[l16][kc + 16 * hi + 8 + i];
        }
        V16 bv;
        bv.h2[0] = *reinterpret_cast<const v8bf*>(vrow + kc + 16 * hi);
        bv.h2[1] = *reinterpret_cast<const v8bf*>(vrow + kc + 16 * hi + 8);
        acc = __builtin_amdgcn_wmma_f32_16x16x32_bf16(false, a.v, false, bv.v,
                                                      (short)0, acc, false, false);
      }
#pragma unroll
      for (int r = 0; r < 8; ++r) {
        const int m = r + 8 * hi;
        const size_t addr = ((size_t)(b * kSn + qbase + m)) * kDn
                          + h * kDK + wv * 16 + l16;
        ctx[addr] = (__bf16)acc[r];
      }
    } else {
      const int t2 = threadIdx.x - 128;
      for (int idx = t2; idx < 16 * kSn; idx += 128) {
        const int q = idx >> 11, col = idx & (kSn - 1);
        const float val = sc[q][col] * (1.0f / 16.0f);
        float* p = attnOut + ((size_t)(b * kSn + qbase + q)) * kSn + col;
        if (h == 0) *p = val; else *p += val;   // owner-thread RMW, replay-safe
      }
    }
    __syncthreads();
  }
}

// ---------------------------------------------------------------------------
// Stage 3: out = ctx(bf16) @ Wo^T + bo, fp32 output (B,S,D).
// Same blocking as proj_kernel; W staged via ASYNCcnt double-buffer pipeline.
// ---------------------------------------------------------------------------
__global__ __launch_bounds__(256)
void outproj_kernel(const __bf16* __restrict__ Xc, const float* __restrict__ W,
                    const float* __restrict__ bias, float* __restrict__ Y)
{
  __shared__ float wl[2][64 * kWP];

  const int tid  = threadIdx.x;
  const int lane = tid & 31;
  const int wv   = tid >> 5;
  const int l16  = lane & 15;
  const int hi   = lane >> 4;
  const int rb   = (blockIdx.x >> 4) * 128;
  const int eb   = (blockIdx.x & 15) * 64;

  const __bf16* xrow = Xc + (size_t)(rb + wv * 16 + l16) * kDn;

  auto stage = [&](int buf, int kk) {
#pragma unroll
    for (int u = 0; u < 2; ++u) {
      const int unit = tid + u * 256;            // 512 x 16B units
      const int row  = unit >> 3;
      const int seg  = unit & 7;
      async_cp16(W + (size_t)(eb + row) * kDn + kk + seg * 4,
                 &wl[buf][row * kWP + seg * 4]);
    }
  };

  v8f acc[4] = {{}, {}, {}, {}};
  stage(0, 0);
  for (int c = 0; c < 32; ++c) {
    const int kk = c * 32;
    if (c + 1 < 32) {
      stage((c + 1) & 1, kk + 32);
      asm volatile("s_wait_asynccnt 0x2" ::: "memory");
    } else {
      asm volatile("s_wait_asynccnt 0x0" ::: "memory");
    }
    __syncthreads();

    V16 a;
    a.h2[0] = *reinterpret_cast<const v8bf*>(xrow + kk + 16 * hi);
    a.h2[1] = *reinterpret_cast<const v8bf*>(xrow + kk + 16 * hi + 8);

    const float* wbuf = wl[c & 1];
#pragma unroll
    for (int nt = 0; nt < 4; ++nt) {
      const float4* q0 =
          reinterpret_cast<const float4*>(wbuf + (nt * 16 + l16) * kWP + 16 * hi);
      V16 bf;
      cvt8(bf, 0, q0[0], q0[1]);
      cvt8(bf, 8, q0[2], q0[3]);
      acc[nt] = __builtin_amdgcn_wmma_f32_16x16x32_bf16(false, a.v, false, bf.v,
                                                        (short)0, acc[nt],
                                                        false, false);
    }
    __syncthreads();
  }

#pragma unroll
  for (int nt = 0; nt < 4; ++nt) {
    const int   e  = eb + nt * 16 + l16;
    const float bn = bias[e];
#pragma unroll
    for (int r = 0; r < 8; ++r)
      Y[(size_t)(rb + wv * 16 + r + 8 * hi) * kDn + e] = acc[nt][r] + bn;
  }
}

// ---------------------------------------------------------------------------
extern "C" void kernel_launch(void* const* d_in, const int* in_sizes, int n_in,
                              void* d_out, int out_size, void* d_ws, size_t ws_size,
                              hipStream_t stream)
{
  const float* query = (const float*)d_in[0];
  const float* key_t = (const float*)d_in[1];
  const float* value = (const float*)d_in[2];
  const float* Wq = (const float*)d_in[3];
  const float* bq = (const float*)d_in[4];
  const float* Wk = (const float*)d_in[5];
  const float* bk = (const float*)d_in[6];
  const float* Wv = (const float*)d_in[7];
  const float* bv = (const float*)d_in[8];
  const float* Wo = (const float*)d_in[9];
  const float* bo = (const float*)d_in[10];

  float* out     = (float*)d_out;
  float* attnOut = out + (size_t)kTok * kDn;     // (B,S,S) region

  __bf16* Qb  = (__bf16*)d_ws;                   // (B,H,S,DK)
  __bf16* Kb  = Qb  + (size_t)kTok * kDn;        // (B,H,S,DK)
  __bf16* Vtb = Kb  + (size_t)kTok * kDn;        // (B,H,DK,S)
  __bf16* Ctx = Vtb + (size_t)kTok * kDn;        // (B,S,D)

  const dim3 blk(256);
  const dim3 gproj(512);                         // 32 row-blocks x 16 e-blocks

  proj_kernel<<<gproj, blk, 0, stream>>>(query, Wq, bq, Qb, 0);
  proj_kernel<<<gproj, blk, 0, stream>>>(key_t, Wk, bk, Kb, 0);
  proj_kernel<<<gproj, blk, 0, stream>>>(value, Wv, bv, Vtb, 1);
  attn_kernel<<<dim3(kBn * (kSn / 16)), blk, 0, stream>>>(Qb, Kb, Vtb, Ctx, attnOut);
  outproj_kernel<<<gproj, blk, 0, stream>>>(Ctx, Wo, bo, out);
}
